// BiMambaBlock_13365938225404
// MI455X (gfx1250) — matrix operand order repaired and checked by reading, not compiled, not run on hardware
//
#include <hip/hip_runtime.h>
#include <hip/hip_bf16.h>
#include <math.h>

#define DIM      512
#define D_STATE  16
#define D_INNER  1024
#define DT_RANK  32
#define B_SZ     2
#define L_SEQ    2048
#define NTOK     (B_SZ * L_SEQ)   // 4096 tokens

#define ASYNC_COPY 1   // use GLOBAL_LOAD_ASYNC_TO_LDS_B128 (ASYNCcnt path)

typedef __attribute__((ext_vector_type(16))) __bf16 v16bf;
typedef __attribute__((ext_vector_type(8)))  __bf16 v8bf;
typedef __attribute__((ext_vector_type(8)))  float  v8f;

__device__ __forceinline__ unsigned short f2bf_rne(float f) {
  unsigned int u = __float_as_uint(f);
  unsigned int lsb = (u >> 16) & 1u;
  u += 0x7fffu + lsb;                 // round-to-nearest-even
  return (unsigned short)(u >> 16);
}

// 16B global -> LDS copy, one per lane. CDNA5 async path bypasses VGPRs.
__device__ __forceinline__ void tile_copy_b128(unsigned short* lptr,
                                               const unsigned short* g) {
#if ASYNC_COPY
  // low 32 bits of the generic pointer to a __shared__ object = LDS byte addr
  unsigned lds_addr = (unsigned)(size_t)lptr;
  asm volatile("global_load_async_to_lds_b128 %0, %1, off"
               :: "v"(lds_addr), "v"(g) : "memory");
#else
  *(uint4*)lptr = *(const uint4*)g;
#endif
}
__device__ __forceinline__ void tile_copy_wait() {
#if ASYNC_COPY
  asm volatile("s_wait_asynccnt 0x0" ::: "memory");
#endif
}

// ---------------------------------------------------------------- converts
__global__ void k_f32_to_bf16(const float* __restrict__ src,
                              unsigned short* __restrict__ dst, int n) {
  int i = blockIdx.x * blockDim.x + threadIdx.x;
  if (i < n) dst[i] = f2bf_rne(src[i]);
}

// ---------------------------------------------------------------- layernorm
__global__ __launch_bounds__(256)
void k_layernorm(const float* __restrict__ x, const float* __restrict__ g,
                 const float* __restrict__ b, float* __restrict__ xn,
                 unsigned short* __restrict__ xn_bf) {
  __shared__ float s1[256], s2[256];
  const int row = blockIdx.x;
  const float* xr = x + (size_t)row * DIM;
  float v0 = xr[threadIdx.x], v1 = xr[threadIdx.x + 256];
  s1[threadIdx.x] = v0 + v1;
  s2[threadIdx.x] = v0 * v0 + v1 * v1;
  __syncthreads();
  for (int s = 128; s > 0; s >>= 1) {
    if (threadIdx.x < s) {
      s1[threadIdx.x] += s1[threadIdx.x + s];
      s2[threadIdx.x] += s2[threadIdx.x + s];
    }
    __syncthreads();
  }
  const float mu  = s1[0] * (1.0f / DIM);
  const float var = s2[0] * (1.0f / DIM) - mu * mu;
  const float inv = rsqrtf(var + 1e-5f);
  float* xo = xn + (size_t)row * DIM;
  unsigned short* xb = xn_bf + (size_t)row * DIM;
  float r0 = (v0 - mu) * inv * g[threadIdx.x]       + b[threadIdx.x];
  float r1 = (v1 - mu) * inv * g[threadIdx.x + 256] + b[threadIdx.x + 256];
  xo[threadIdx.x]       = r0;  xb[threadIdx.x]       = f2bf_rne(r0);
  xo[threadIdx.x + 256] = r1;  xb[threadIdx.x + 256] = f2bf_rne(r1);
}

// ---------------------------------------------------------------- WMMA GEMM
// C[M,N] = act( A[M,K](bf16, stride lda, optional per-batch row reversal)
//               @ W[N,K]^T (bf16) + bias[N] + resid[M,N] );  bf16 C copy optional.
// Block: 256 thr = 8 waves. BN=128: wave = 16Mx64N (4 accs). BN=64: 16Mx32N.
// Requires M%64==0, N%BN==0, K%32==0.
template<int BN>
__global__ __launch_bounds__(256)
void k_gemm_bf16(const unsigned short* __restrict__ Abf, int lda,
                 const unsigned short* __restrict__ W, int K,
                 float* __restrict__ C, int ldc,
                 unsigned short* __restrict__ Cbf,
                 const float* __restrict__ bias,
                 const float* __restrict__ resid,
                 int act,      // 0: none, 1: softplus
                 int revL)     // 0: normal; else L for per-batch time reversal of A reads
{
  constexpr int NACC = BN / 32;          // 16x16 accumulators per wave
  constexpr int WCH  = (BN * 4) / 256;   // W-tile 16B chunks per thread
  __shared__ unsigned short As[64][40];  // rows padded to 40 shorts -> conflict-free b128
  __shared__ unsigned short Ws[BN][40];

  const int tid = threadIdx.x;
  const int m0  = blockIdx.y * 64;
  const int n0  = blockIdx.x * BN;

  const int ar = tid >> 2;
  const int ac = (tid & 3) * 8;
  int arow = m0 + ar;
  if (revL) { int bb = arow / revL; int t = arow - bb * revL; arow = bb * revL + (revL - 1 - t); }
  const unsigned short* Arow = Abf + (size_t)arow * lda;

  const int lane = tid & 31;
  const int wave = tid >> 5;
  const int wm   = wave >> 1;     // 0..3  (M tile)
  const int wn   = wave & 1;      // 0..1  (N half)
  const int ln   = lane & 15;
  const int hi   = lane >> 4;

  v8f acc[NACC] = {};

  for (int k0 = 0; k0 < K; k0 += 32) {
    __syncthreads();
    // A tile: 64 x 32 bf16, one async b128 per thread
    tile_copy_b128(&As[ar][ac], Arow + k0 + ac);
    // W tile: BN x 32 bf16
    #pragma unroll
    for (int it = 0; it < WCH; ++it) {
      const int q = tid + it * 256;
      const int r = q >> 2, cc = (q & 3) * 8;
      tile_copy_b128(&Ws[r][cc], W + (size_t)(n0 + r) * K + k0 + cc);
    }
    if (k0 + 32 < K) __builtin_prefetch(Arow + k0 + 32 + ac, 0, 1);  // global_prefetch_b8
    tile_copy_wait();          // s_wait_asynccnt 0
    __syncthreads();

    // A fragment: 16x32 bf16. elem i -> K = (i>>3)*16 + hi*8 + (i&7)
    v16bf a;
    {
      const v8bf alo = *(const v8bf*)&As[wm * 16 + ln][hi * 8];
      const v8bf ahi = *(const v8bf*)&As[wm * 16 + ln][16 + hi * 8];
      #pragma unroll
      for (int i = 0; i < 8; ++i) { a[i] = alo[i]; a[8 + i] = ahi[i]; }
    }
    // B fragments: 32x16 bf16. elem i -> K = (i>>3)*16 + hi*8 + (i&7) ; Ws row is N index.
    #pragma unroll
    for (int j = 0; j < NACC; ++j) {
      const int brow = wn * (BN / 2) + j * 16 + ln;
      const v8bf blo = *(const v8bf*)&Ws[brow][hi * 8];
      const v8bf bhi = *(const v8bf*)&Ws[brow][16 + hi * 8];
      v16bf b;
      #pragma unroll
      for (int i = 0; i < 8; ++i) { b[i] = blo[i]; b[8 + i] = bhi[i]; }
      acc[j] = __builtin_amdgcn_wmma_f32_16x16x32_bf16(false, a, false, b,
                                                       (short)0, acc[j], false, false);
    }
  }

  // epilogue: D layout -> M = hi*8 + r, N = ln
  const int outm = m0 + wm * 16 + hi * 8;
  #pragma unroll
  for (int j = 0; j < NACC; ++j) {
    const int col = n0 + wn * (BN / 2) + j * 16 + ln;
    const float bv = bias ? bias[col] : 0.0f;
    #pragma unroll
    for (int r = 0; r < 8; ++r) {
      const int row = outm + r;
      float v = acc[j][r] + bv;
      if (resid) v += resid[(size_t)row * ldc + col];
      if (act == 1) v = (v > 20.0f) ? v : log1pf(__expf(v));
      C[(size_t)row * ldc + col] = v;
      if (Cbf) Cbf[(size_t)row * ldc + col] = f2bf_rne(v);
    }
  }
}

// ---------------------------------------------------------------- conv + SiLU
__global__ __launch_bounds__(256)
void k_conv_silu(const float* __restrict__ xz,     // (NTOK, 2048); xc = cols [0,1024)
                 const float* __restrict__ convw,  // (1024,4)
                 const float* __restrict__ convb,  // (1024,)
                 float* __restrict__ xconv,        // (NTOK, 1024)
                 unsigned short* __restrict__ xconv_bf)
{
  const int idx = blockIdx.x * blockDim.x + threadIdx.x;   // NTOK*1024
  const int c   = idx & (D_INNER - 1);
  const int row = idx >> 10;
  const int b   = row >> 11;            // / L_SEQ
  const int t   = row & (L_SEQ - 1);
  const float w0 = convw[c * 4 + 0], w1 = convw[c * 4 + 1];
  const float w2 = convw[c * 4 + 2], w3 = convw[c * 4 + 3];
  const float* base = xz + (size_t)b * L_SEQ * (2 * D_INNER) + c;
  float acc = convb[c];
  if (t >= 3) acc += w0 * base[(size_t)(t - 3) * (2 * D_INNER)];
  if (t >= 2) acc += w1 * base[(size_t)(t - 2) * (2 * D_INNER)];
  if (t >= 1) acc += w2 * base[(size_t)(t - 1) * (2 * D_INNER)];
  acc += w3 * base[(size_t)t * (2 * D_INNER)];
  acc = acc / (1.0f + __expf(-acc));
  xconv[idx] = acc;
  xconv_bf[idx] = f2bf_rne(acc);
}

// ---------------------------------------------------------------- selective scan
#define SCAN_T 64
__global__ __launch_bounds__(256)
void k_scan(const float* __restrict__ delta,  // (NTOK,1024)
            const float* __restrict__ dbc,    // (NTOK,64): B=cols32..47, C=48..63
            const float* __restrict__ xconv,  // (NTOK,1024)
            const float* __restrict__ xz,     // (NTOK,2048): z = cols 1024..2047
            const float* __restrict__ Alog,   // (1024,16)
            const float* __restrict__ Dp,     // (1024,)
            float* __restrict__ ys,           // (NTOK,1024)
            unsigned short* __restrict__ ys_bf)
{
  __shared__ float bc[SCAN_T][32];
  const int tid = threadIdx.x;
  const int b   = blockIdx.x >> 2;
  const int c   = ((blockIdx.x & 3) << 8) + tid;

  float An[D_STATE], h[D_STATE];
  #pragma unroll
  for (int n = 0; n < D_STATE; ++n) { An[n] = -__expf(Alog[c * D_STATE + n]); h[n] = 0.0f; }
  const float Dc = Dp[c];

  for (int t0 = 0; t0 < L_SEQ; t0 += SCAN_T) {
    __syncthreads();
    { // stage B,C for 64 timesteps
      const int r  = tid >> 2;
      const int i0 = (tid & 3) * 8;
      const float* src = dbc + ((size_t)(b * L_SEQ + t0 + r)) * 64 + 32 + i0;
      *(float4*)&bc[r][i0]     = *(const float4*)src;
      *(float4*)&bc[r][i0 + 4] = *(const float4*)(src + 4);
    }
    __syncthreads();
    for (int tt = 0; tt < SCAN_T; ++tt) {
      const size_t row = (size_t)b * L_SEQ + t0 + tt;
      const float dv  = delta[row * D_INNER + c];
      const float xcv = xconv[row * D_INNER + c];
      const float zv  = xz[row * (2 * D_INNER) + D_INNER + c];
      const float dx  = dv * xcv;
      float y = 0.0f;
      #pragma unroll
      for (int n = 0; n < D_STATE; ++n) {
        const float dA = __expf(dv * An[n]);
        h[n] = dA * h[n] + dx * bc[tt][n];
        y += h[n] * bc[tt][16 + n];
      }
      const float sz = zv / (1.0f + __expf(-zv));
      const float yo = (y + xcv * Dc) * sz;
      ys[row * D_INNER + c] = yo;
      ys_bf[row * D_INNER + c] = f2bf_rne(yo);
    }
  }
}

// ---------------------------------------------------------------- combine dirs
__global__ __launch_bounds__(256)
void k_combine(const float* __restrict__ of, const float* __restrict__ ob,
               float* __restrict__ s, unsigned short* __restrict__ s_bf) {
  const int idx = blockIdx.x * blockDim.x + threadIdx.x;  // NTOK*512
  const int c   = idx & (DIM - 1);
  const int row = idx >> 9;
  const int b   = row >> 11;
  const int t   = row & (L_SEQ - 1);
  const int rr  = b * L_SEQ + (L_SEQ - 1 - t);
  const float v = of[idx] + ob[(size_t)rr * DIM + c];
  s[idx] = v;
  s_bf[idx] = f2bf_rne(v);
}

// ---------------------------------------------------------------- host launch
extern "C" void kernel_launch(void* const* d_in, const int* in_sizes, int n_in,
                              void* d_out, int out_size, void* d_ws, size_t ws_size,
                              hipStream_t stream) {
  const float* x      = (const float*)d_in[0];
  const float* ln_g   = (const float*)d_in[1];
  const float* ln_b   = (const float*)d_in[2];
  const float* fuse_w = (const float*)d_in[21];
  const float* fuse_b = (const float*)d_in[22];
  float* out = (float*)d_out;

  size_t o = 0;
  auto alloc = [&](size_t bytes) { size_t r = o; o = (o + bytes + 255) & ~(size_t)255; return r; };
  const size_t oXN    = alloc((size_t)NTOK * DIM * 4);
  const size_t oXNBF  = alloc((size_t)NTOK * DIM * 2);
  const size_t oXZ    = alloc((size_t)NTOK * 2 * D_INNER * 4);
  const size_t oXCONV = alloc((size_t)NTOK * D_INNER * 4);
  const size_t oXCBF  = alloc((size_t)NTOK * D_INNER * 2);
  const size_t oDBC   = alloc((size_t)NTOK * 64 * 4);
  const size_t oDBCBF = alloc((size_t)NTOK * 64 * 2);
  const size_t oDELTA = alloc((size_t)NTOK * D_INNER * 4);
  const size_t oYS    = alloc((size_t)NTOK * D_INNER * 4);
  const size_t oYSBF  = alloc((size_t)NTOK * D_INNER * 2);
  const size_t oOUTF  = alloc((size_t)NTOK * DIM * 4);
  const size_t oOUTB  = alloc((size_t)NTOK * DIM * 4);
  const size_t oS     = alloc((size_t)NTOK * DIM * 4);
  const size_t oSBF   = alloc((size_t)NTOK * DIM * 2);
  const size_t oWIN   = alloc((size_t)2 * D_INNER * DIM * 2);
  const size_t oWXP   = alloc((size_t)64 * D_INNER * 2);
  const size_t oWDT   = alloc((size_t)D_INNER * DT_RANK * 2);
  const size_t oWOUT  = alloc((size_t)DIM * D_INNER * 2);
  const size_t oWFUSE = alloc((size_t)DIM * DIM * 2);
  if (ws_size < o) return;

  char* ws = (char*)d_ws;
  float* xn    = (float*)(ws + oXN);
  float* xz    = (float*)(ws + oXZ);
  float* xconv = (float*)(ws + oXCONV);
  float* dbc   = (float*)(ws + oDBC);
  float* delta = (float*)(ws + oDELTA);
  float* ys    = (float*)(ws + oYS);
  float* outF  = (float*)(ws + oOUTF);
  float* outB  = (float*)(ws + oOUTB);
  float* sbuf  = (float*)(ws + oS);
  unsigned short* xn_bf    = (unsigned short*)(ws + oXNBF);
  unsigned short* xconv_bf = (unsigned short*)(ws + oXCBF);
  unsigned short* dbc_bf   = (unsigned short*)(ws + oDBCBF);
  unsigned short* ys_bf    = (unsigned short*)(ws + oYSBF);
  unsigned short* s_bf     = (unsigned short*)(ws + oSBF);
  unsigned short* wIn   = (unsigned short*)(ws + oWIN);
  unsigned short* wXp   = (unsigned short*)(ws + oWXP);
  unsigned short* wDt   = (unsigned short*)(ws + oWDT);
  unsigned short* wOut  = (unsigned short*)(ws + oWOUT);
  unsigned short* wFuse = (unsigned short*)(ws + oWFUSE);

  auto cvt = [&](const float* src, unsigned short* dst, int n) {
    k_f32_to_bf16<<<(n + 255) / 256, 256, 0, stream>>>(src, dst, n);
  };
  auto gemm = [&](const unsigned short* A, int lda, const unsigned short* W, int K,
                  float* C, int ldc, int M, int N, unsigned short* Cbf,
                  const float* bias, const float* resid, int act, int revL) {
    if (N % 128 == 0) {
      dim3 grid(N / 128, M / 64);
      k_gemm_bf16<128><<<grid, 256, 0, stream>>>(A, lda, W, K, C, ldc, Cbf, bias, resid, act, revL);
    } else {
      dim3 grid(N / 64, M / 64);
      k_gemm_bf16<64><<<grid, 256, 0, stream>>>(A, lda, W, K, C, ldc, Cbf, bias, resid, act, revL);
    }
  };

  k_layernorm<<<NTOK, 256, 0, stream>>>(x, ln_g, ln_b, xn, xn_bf);

  auto run_dir = [&](int pbase, float* out_dir, int rev) {
    const float* convw  = (const float*)d_in[pbase + 1];
    const float* convb  = (const float*)d_in[pbase + 2];
    const float* dtb    = (const float*)d_in[pbase + 5];
    const float* Alog   = (const float*)d_in[pbase + 6];
    const float* Dp     = (const float*)d_in[pbase + 7];

    cvt((const float*)d_in[pbase + 0], wIn,  2 * D_INNER * DIM);  // inproj
    cvt((const float*)d_in[pbase + 3], wXp,  64 * D_INNER);       // xproj
    cvt((const float*)d_in[pbase + 4], wDt,  D_INNER * DT_RANK);  // dtw
    cvt((const float*)d_in[pbase + 8], wOut, DIM * D_INNER);      // outw

    // xz = xn(rev?) @ inproj^T                       (4096 x 2048, K=512)
    gemm(xn_bf, DIM, wIn, DIM, xz, 2 * D_INNER, NTOK, 2 * D_INNER,
         nullptr, nullptr, nullptr, 0, rev ? L_SEQ : 0);
    // depthwise causal conv + SiLU
    k_conv_silu<<<(NTOK * D_INNER) / 256, 256, 0, stream>>>(xz, convw, convb, xconv, xconv_bf);
    // dbc = xconv @ xproj^T                          (4096 x 64, K=1024) + bf16 copy
    gemm(xconv_bf, D_INNER, wXp, D_INNER, dbc, 64, NTOK, 64, dbc_bf, nullptr, nullptr, 0, 0);
    // delta = softplus(dbc[:, :32] @ dtw^T + dtb)    (4096 x 1024, K=32; lda=64)
    gemm(dbc_bf, 64, wDt, DT_RANK, delta, D_INNER, NTOK, D_INNER, nullptr, dtb, nullptr, 1, 0);
    // selective scan (+ D skip + SiLU(z) gate)
    k_scan<<<B_SZ * (D_INNER / 256), 256, 0, stream>>>(delta, dbc, xconv, xz, Alog, Dp, ys, ys_bf);
    // out = ys @ outw^T                              (4096 x 512, K=1024)
    gemm(ys_bf, D_INNER, wOut, D_INNER, out_dir, DIM, NTOK, DIM, nullptr, nullptr, nullptr, 0, 0);
  };

  run_dir(3, outF, 0);    // forward
  run_dir(12, outB, 1);   // backward (time-reversed end-to-end)

  k_combine<<<(NTOK * DIM) / 256, 256, 0, stream>>>(outF, outB, sbuf, s_bf);

  cvt(fuse_w, wFuse, DIM * DIM);
  // out = s @ fuse_w^T + fuse_b + x
  gemm(s_bf, DIM, wFuse, DIM, out, DIM, NTOK, DIM, nullptr, fuse_b, x, 0, 0);
}
